// TitansTNT_32134945309319
// MI455X (gfx1250) — matrix operand order repaired and checked by reading, not compiled, not run on hardware
//
#include <hip/hip_runtime.h>
#include <hip/hip_bf16.h>
#include <math.h>

// ---------------- problem constants (from reference) ----------------
#define DD 1024
#define HH 4
#define HD 256
#define WIN 128
#define NPMEM 4
#define BB 2
#define LL 4096
#define LT (LL + NPMEM)       // 4100
#define LTP 4128              // padded key stride for transposed V
#define QTILES ((LT + 15) / 16)  // 257

typedef __bf16 bf16;
typedef __attribute__((ext_vector_type(16))) __bf16 v16bf;
typedef __attribute__((ext_vector_type(8)))  __bf16 v8bf;
typedef __attribute__((ext_vector_type(8)))  float  v8f;

// ---- async global->LDS path (CDNA5), with graceful fallback ----
#if defined(__has_builtin)
#if __has_builtin(__builtin_amdgcn_global_load_async_to_lds_b128) && \
    __has_builtin(__builtin_amdgcn_s_wait_asynccnt)
#define USE_ASYNC_LDS 1
#endif
#endif

#ifdef USE_ASYNC_LDS
// builtin signature (from clang diagnostic): param0 = v4i addrspace(1)*,
// param1 = v4i addrspace(3)*, then two constant ints (offset, cpol).
typedef int v4i_ __attribute__((vector_size(16)));
typedef __attribute__((address_space(1))) v4i_ gv4i;
typedef __attribute__((address_space(3))) v4i_ lv4i;
#endif

__device__ __forceinline__ v8bf load8(const bf16* p) {
  return *reinterpret_cast<const v8bf*>(p);
}
__device__ __forceinline__ v16bf join16(v8bf lo, v8bf hi) {
  v16bf r;
#pragma unroll
  for (int i = 0; i < 8; ++i) { r[i] = lo[i]; r[8 + i] = hi[i]; }
  return r;
}
__device__ __forceinline__ v8f vzero8() {
  v8f z = {0.f, 0.f, 0.f, 0.f, 0.f, 0.f, 0.f, 0.f};
  return z;
}

// ====================================================================
// Generic bf16 GEMM, f32 accumulate via V_WMMA_F32_16X16X32_BF16.
// Block tile 128(M) x 128(N); 8 waves as 4(M) x 2(N); each wave owns
// 32x64 = 2x4 WMMA tiles. K step 32, double-buffered LDS.
// A-panel staged with GLOBAL_LOAD_ASYNC_TO_LDS_B128 (ASYNCcnt) when the
// toolchain exposes the builtin; B-panel goes through registers because
// it is transposed on the way into LDS.
// MODE 0: out f32 = acc
// MODE 2: gate    : g = sigmoid(acc + bias[n]); out_bf = aux_bf[idx] * g
// MODE 3: silu    : t = acc + bias[n]; out_bf = t*sigmoid(t)
// MODE 4: out f32 = acc + bias[n] + aux_f32[idx]
// MODE 5: out f32 = acc + bias[n]
// ====================================================================
template <int MODE>
__global__ __launch_bounds__(256) void gemm_bf16_kernel(
    const bf16* __restrict__ A, const bf16* __restrict__ Bw,
    const float* __restrict__ bias, const void* __restrict__ aux,
    void* __restrict__ outp, int M, int N, int K) {
  __shared__ bf16 As[2][128][40];   // 80B row stride: conflict-free, 16B aligned
  __shared__ bf16 Bs[2][128][40];   // transposed B tile: Bs[buf][n][k]

  const int tid  = threadIdx.x;
  const int wave = tid >> 5;
  const int lane = tid & 31;
  const int wm = (wave >> 1) * 32;    // wave M offset in tile
  const int wn = (wave & 1) * 64;     // wave N offset in tile
  const int m0 = blockIdx.y * 128;
  const int n0 = blockIdx.x * 128;

  const int r   = lane & 15;
  const int kb2 = (lane >> 4) * 8;    // A-frag K split
  const int kb3 = (lane >> 4) * 16;   // unused; B-frag uses kb2 like A

  v8f acc[2][4];
#pragma unroll
  for (int i = 0; i < 2; ++i)
#pragma unroll
    for (int t = 0; t < 4; ++t) acc[i][t] = vzero8();

  const int ar  = tid >> 1;           // 0..127 : A row in tile
  const int ac  = (tid & 1) * 16;     // 0/16   : A col segment
  const int bkr = tid >> 3;           // 0..31  : B k-row
  const int bc  = (tid & 7) * 16;     // 0..112 : B col segment

  int4 a0 = {0, 0, 0, 0}, a1 = {0, 0, 0, 0};
  int4 b0 = {0, 0, 0, 0}, b1 = {0, 0, 0, 0};

  // ---- A staging ----
#ifdef USE_ASYNC_LDS
  auto stage_A = [&](int buf, int k0) {
    const int gm = m0 + ar;
    if (gm < M) {
      const bf16* src = A + (size_t)gm * K + k0 + ac;
      __builtin_amdgcn_global_load_async_to_lds_b128(
          (gv4i*)src, (lv4i*)&As[buf][ar][ac], 0, 0);
      __builtin_amdgcn_global_load_async_to_lds_b128(
          (gv4i*)(src + 8), (lv4i*)&As[buf][ar][ac + 8], 0, 0);
    } else {
      *reinterpret_cast<int4*>(&As[buf][ar][ac])     = int4{0, 0, 0, 0};
      *reinterpret_cast<int4*>(&As[buf][ar][ac + 8]) = int4{0, 0, 0, 0};
    }
  };
#else
  auto fetch_A = [&](int k0) {
    const int gm = m0 + ar;
    if (gm < M) {
      const int4* src = reinterpret_cast<const int4*>(A + (size_t)gm * K + k0 + ac);
      a0 = src[0];
      a1 = src[1];
    } else {
      a0 = int4{0, 0, 0, 0};
      a1 = int4{0, 0, 0, 0};
    }
  };
  auto store_A = [&](int buf) {
    *reinterpret_cast<int4*>(&As[buf][ar][ac])     = a0;
    *reinterpret_cast<int4*>(&As[buf][ar][ac + 8]) = a1;
  };
#endif
  // ---- B staging (register transpose) ----
  auto fetch_B = [&](int k0) {
    const int4* bsrc = reinterpret_cast<const int4*>(Bw + (size_t)(k0 + bkr) * N + n0 + bc);
    b0 = bsrc[0];
    b1 = bsrc[1];
  };
  auto store_B = [&](int buf) {
    const bf16* bv0 = reinterpret_cast<const bf16*>(&b0);
    const bf16* bv1 = reinterpret_cast<const bf16*>(&b1);
#pragma unroll
    for (int i = 0; i < 8; ++i) Bs[buf][bc + i][bkr]     = bv0[i];
#pragma unroll
    for (int i = 0; i < 8; ++i) Bs[buf][bc + 8 + i][bkr] = bv1[i];
  };

  const int nk = K >> 5;
  // prologue: stage K-step 0 into buffer 0
#ifdef USE_ASYNC_LDS
  stage_A(0, 0);
#else
  fetch_A(0);
  store_A(0);
#endif
  fetch_B(0);
  store_B(0);
#ifdef USE_ASYNC_LDS
  __builtin_amdgcn_s_wait_asynccnt(0);
#endif
  __syncthreads();

  for (int t = 0; t < nk; ++t) {
    const int cur = t & 1;
    const bool more = (t + 1 < nk);
    if (more) {
#ifdef USE_ASYNC_LDS
      stage_A(cur ^ 1, (t + 1) << 5);   // DMA straight into back buffer
#else
      fetch_A((t + 1) << 5);
#endif
      fetch_B((t + 1) << 5);
    }

    const v16bf af0 = join16(load8(&As[cur][wm + r][kb2]),
                             load8(&As[cur][wm + r][16 + kb2]));
    const v16bf af1 = join16(load8(&As[cur][wm + 16 + r][kb2]),
                             load8(&As[cur][wm + 16 + r][16 + kb2]));
#pragma unroll
    for (int nt = 0; nt < 4; ++nt) {
      const bf16* bp = &Bs[cur][wn + nt * 16 + r][kb2];
      const v16bf bf_ = join16(load8(bp), load8(bp + 16));
      acc[0][nt] = __builtin_amdgcn_wmma_f32_16x16x32_bf16(
          false, af0, false, bf_, (short)0, acc[0][nt], false, false);
      acc[1][nt] = __builtin_amdgcn_wmma_f32_16x16x32_bf16(
          false, af1, false, bf_, (short)0, acc[1][nt], false, false);
    }

    if (more) {
#ifndef USE_ASYNC_LDS
      store_A(cur ^ 1);
#endif
      store_B(cur ^ 1);
#ifdef USE_ASYNC_LDS
      __builtin_amdgcn_s_wait_asynccnt(0);
#endif
    }
    __syncthreads();
  }

  // ---- epilogue (C layout: row m = (lane>>4)*8 + j, col n = lane&15) ----
#pragma unroll
  for (int i = 0; i < 2; ++i) {
#pragma unroll
    for (int nt = 0; nt < 4; ++nt) {
#pragma unroll
      for (int j = 0; j < 8; ++j) {
        const int m = m0 + wm + i * 16 + (lane >> 4) * 8 + j;
        const int n = n0 + wn + nt * 16 + (lane & 15);
        if (m >= M) continue;
        const float a = acc[i][nt][j];
        const size_t idx = (size_t)m * N + n;
        if (MODE == 0) {
          ((float*)outp)[idx] = a;
        } else if (MODE == 2) {
          const float g = 1.0f / (1.0f + __expf(-(a + bias[n])));
          ((bf16*)outp)[idx] = (bf16)((float)((const bf16*)aux)[idx] * g);
        } else if (MODE == 3) {
          const float t2 = a + bias[n];
          ((bf16*)outp)[idx] = (bf16)(t2 / (1.0f + __expf(-t2)));
        } else if (MODE == 4) {
          ((float*)outp)[idx] = a + bias[n] + ((const float*)aux)[idx];
        } else if (MODE == 5) {
          ((float*)outp)[idx] = a + bias[n];
        }
      }
    }
  }
}

// ====================================================================
// Banded flash attention. One wave per 16-query tile per (b,h).
// Q,K in (B,H,Lt,HD) bf16; V transposed in (B,H,HD,LTP) bf16.
// Output attn (B,Lt,D=H*HD) bf16.
// ====================================================================
__global__ __launch_bounds__(256) void attn_kernel(
    const bf16* __restrict__ qm, const bf16* __restrict__ km,
    const bf16* __restrict__ vt, bf16* __restrict__ outp) {
  __shared__ bf16 plds[8][16][32];   // per-wave P transpose tile

  const int wave = threadIdx.x >> 5;
  const int lane = threadIdx.x & 31;
  const int task = blockIdx.x * 8 + wave;
  const int total = BB * HH * QTILES;
  if (task >= total) return;
  const int qt = task % QTILES;
  const int h  = (task / QTILES) % HH;
  const int b  = task / (QTILES * HH);

  const size_t headqk = (size_t)(b * HH + h) * LT * HD;
  const size_t headv  = (size_t)(b * HH + h) * HD * (size_t)LTP;
  const int r   = lane & 15;
  const int kb2 = (lane >> 4) * 8;
  const int kb3 = (lane >> 4) * 16;

  // hoist 16x256 Q tile as 8 A-fragments (K=32 each)
  v16bf aq[8];
  {
    int qrow = qt * 16 + r;
    if (qrow >= LT) qrow = LT - 1;
    const bf16* qp = qm + headqk + (size_t)qrow * HD;
#pragma unroll
    for (int c = 0; c < 8; ++c) {
      const bf16* p = qp + c * 32 + kb2;
      aq[c] = join16(load8(p), load8(p + 16));
    }
  }

  v8f o[16];
#pragma unroll
  for (int t = 0; t < 16; ++t) o[t] = vzero8();
  float rowmax[8], rowsum[8];
#pragma unroll
  for (int j = 0; j < 8; ++j) { rowmax[j] = -1e30f; rowsum[j] = 0.0f; }

  int kstart = qt * 16 - WIN;
  if (kstart < 0) kstart = 0;
  kstart &= ~31;
  const int qhi = qt * 16 + 15;
  const int kend = (qhi < LT - 1) ? qhi : (LT - 1);

  for (int kb = kstart; kb <= kend; kb += 32) {
    // ---- S = Q K^T for 32 keys (two 16x16 score tiles) ----
    v8f sc0 = vzero8(), sc1 = vzero8();
    {
      int key0 = kb + r;       if (key0 > LT - 1) key0 = LT - 1;
      int key1 = kb + 16 + r;  if (key1 > LT - 1) key1 = LT - 1;
      const bf16* kp0 = km + headqk + (size_t)key0 * HD;
      const bf16* kp1 = km + headqk + (size_t)key1 * HD;
#pragma unroll
      for (int c = 0; c < 8; ++c) {
        const bf16* p0 = kp0 + c * 32 + kb2;
        const bf16* p1 = kp1 + c * 32 + kb2;
        const v16bf bk0 = join16(load8(p0), load8(p0 + 16));
        const v16bf bk1 = join16(load8(p1), load8(p1 + 16));
        sc0 = __builtin_amdgcn_wmma_f32_16x16x32_bf16(false, aq[c], false, bk0, (short)0, sc0, false, false);
        sc1 = __builtin_amdgcn_wmma_f32_16x16x32_bf16(false, aq[c], false, bk1, (short)0, sc1, false, false);
      }
    }
    // ---- scale, band-mask, online softmax ----
    float p0v[8], p1v[8], factor[8];
#pragma unroll
    for (int j = 0; j < 8; ++j) {
      const int m  = qt * 16 + (lane >> 4) * 8 + j;
      const int n0 = kb + r;
      const int n1 = kb + 16 + r;
      float s0 = sc0[j] * 0.0625f;   // 1/sqrt(HD=256)
      float s1 = sc1[j] * 0.0625f;
      if (!((n0 <= m) && (n0 >= m - WIN) && (m < LT))) s0 = -1e30f;
      if (!((n1 <= m) && (n1 >= m - WIN) && (m < LT))) s1 = -1e30f;
      float mj = fmaxf(s0, s1);
#pragma unroll
      for (int d = 1; d < 16; d <<= 1) mj = fmaxf(mj, __shfl_xor(mj, d, 32));
      const float newmax = fmaxf(rowmax[j], mj);
      factor[j] = __expf(rowmax[j] - newmax);
      rowmax[j] = newmax;
      const float e0 = __expf(s0 - newmax);
      const float e1 = __expf(s1 - newmax);
      float ps = e0 + e1;
#pragma unroll
      for (int d = 1; d < 16; d <<= 1) ps += __shfl_xor(ps, d, 32);
      rowsum[j] = rowsum[j] * factor[j] + ps;
      p0v[j] = e0;
      p1v[j] = e1;
    }
    // rescale accumulators
#pragma unroll
    for (int t = 0; t < 16; ++t)
#pragma unroll
      for (int j = 0; j < 8; ++j) o[t][j] *= factor[j];

    // ---- transpose P (C-layout -> A-layout) through per-wave LDS ----
#pragma unroll
    for (int j = 0; j < 8; ++j) {
      plds[wave][(lane >> 4) * 8 + j][r]      = (bf16)p0v[j];
      plds[wave][(lane >> 4) * 8 + j][16 + r] = (bf16)p1v[j];
    }
    asm volatile("s_wait_dscnt 0" ::: "memory");
    const v16bf pf = join16(load8(&plds[wave][r][kb2]),
                            load8(&plds[wave][r][16 + kb2]));
    // ---- O += P V (16 n-tiles over HD=256) ----
#pragma unroll
    for (int t = 0; t < 16; ++t) {
      const bf16* vp = vt + headv + (size_t)(t * 16 + r) * LTP + kb + kb2;
      const v16bf bv = join16(load8(vp), load8(vp + 16));
      o[t] = __builtin_amdgcn_wmma_f32_16x16x32_bf16(false, pf, false, bv, (short)0, o[t], false, false);
    }
  }

  // ---- normalize and store to (B,Lt,D) bf16 ----
#pragma unroll
  for (int t = 0; t < 16; ++t) {
#pragma unroll
    for (int j = 0; j < 8; ++j) {
      const int m = qt * 16 + (lane >> 4) * 8 + j;
      if (m < LT) {
        const float val = o[t][j] / rowsum[j];
        outp[((size_t)b * LT + m) * DD + h * HD + t * 16 + (lane & 15)] = (bf16)val;
      }
    }
  }
}

// ====================================================================
// Elementwise / prep kernels
// ====================================================================
__global__ void cvt_bf16_kernel(const float* __restrict__ in, bf16* __restrict__ out, size_t n) {
  const size_t i = (size_t)blockIdx.x * blockDim.x + threadIdx.x;
  if (i < n) out[i] = (bf16)in[i];
}

__global__ void build_xp_kernel(const float* __restrict__ x, const float* __restrict__ pmem,
                                bf16* __restrict__ xp) {
  const size_t i = (size_t)blockIdx.x * blockDim.x + threadIdx.x;
  const size_t n = (size_t)BB * LT * DD;
  if (i >= n) return;
  const int d = (int)(i % DD);
  const size_t row = i / DD;
  const int l = (int)(row % LT);
  const int b = (int)(row / LT);
  const float v = (l < NPMEM) ? pmem[l * DD + d]
                              : x[((size_t)b * LL + (l - NPMEM)) * DD + d];
  xp[i] = (bf16)v;
}

// rotary: in (B,Lt,D) f32 -> out (B,H,Lt,HD) bf16
__global__ void rope_kernel(const float* __restrict__ t, const float* __restrict__ cosp,
                            const float* __restrict__ sinp, bf16* __restrict__ out) {
  const size_t i = (size_t)blockIdx.x * blockDim.x + threadIdx.x;
  const size_t n = (size_t)BB * LT * DD;
  if (i >= n) return;
  const int d = (int)(i % HD);
  const size_t r = i / HD;
  const int hh = (int)(r % HH);
  const size_t row = r / HH;            // b*LT + l
  const int l = (int)(row % LT);
  const int b = (int)(row / LT);
  const float* base = t + row * DD + hh * HD;
  const float v = base[d];
  const float rot = (d < HD / 2) ? -base[d + HD / 2] : base[d - HD / 2];
  const float o = v * cosp[(size_t)l * HD + d] + rot * sinp[(size_t)l * HD + d];
  out[((size_t)(b * HH + hh) * LT + l) * HD + d] = (bf16)o;
}

// V: (B,Lt,D) f32 -> transposed (B,H,HD,LTP) bf16
__global__ void vtrans_kernel(const float* __restrict__ t, bf16* __restrict__ out) {
  const size_t i = (size_t)blockIdx.x * blockDim.x + threadIdx.x;
  const size_t n = (size_t)BB * LT * DD;
  if (i >= n) return;
  const int d = (int)(i % HD);
  const size_t r = i / HD;
  const int hh = (int)(r % HH);
  const size_t row = r / HH;
  const int l = (int)(row % LT);
  const int b = (int)(row / LT);
  out[((size_t)(b * HH + hh) * HD + d) * LTP + l] = (bf16)t[row * DD + hh * HD + d];
}

// combined (B,L,3D) bf16 = [global_out | local_out | core(rows +NPMEM)]
__global__ void pack_combined_kernel(const float* __restrict__ gout, const float* __restrict__ lout,
                                     const float* __restrict__ core, bf16* __restrict__ comb) {
  const size_t i = (size_t)blockIdx.x * blockDim.x + threadIdx.x;
  const size_t n = (size_t)BB * LL * 3 * DD;
  if (i >= n) return;
  const int c = (int)(i % (3 * DD));
  const size_t row = i / (3 * DD);      // b*LL + l
  const int l = (int)(row % LL);
  const int b = (int)(row / LL);
  float v;
  if (c < DD)            v = gout[row * DD + c];
  else if (c < 2 * DD)   v = lout[row * DD + (c - DD)];
  else                   v = core[((size_t)b * LT + l + NPMEM) * DD + (c - 2 * DD)];
  comb[i] = (bf16)v;
}

__device__ __forceinline__ float block_reduce_sum(float v, float* sbuf) {
  sbuf[threadIdx.x] = v;
  __syncthreads();
  for (int st = 128; st > 0; st >>= 1) {
    if ((int)threadIdx.x < st) sbuf[threadIdx.x] += sbuf[threadIdx.x + st];
    __syncthreads();
  }
  const float r = sbuf[0];
  __syncthreads();
  return r;
}

// x1 = resid + LN(mo)*g + b   (f32 out)
__global__ __launch_bounds__(256) void ln_add_kernel(
    const float* __restrict__ mo, const float* __restrict__ resid,
    const float* __restrict__ g, const float* __restrict__ bt, float* __restrict__ x1) {
  __shared__ float sbuf[256];
  const size_t row = blockIdx.x;
  const float* p = mo + row * DD;
  float s = 0.f, sq = 0.f;
  for (int i = threadIdx.x; i < DD; i += 256) { const float v = p[i]; s += v; sq += v * v; }
  const float mean = block_reduce_sum(s, sbuf) * (1.0f / DD);
  const float var  = block_reduce_sum(sq, sbuf) * (1.0f / DD) - mean * mean;
  const float rstd = rsqrtf(var + 1e-5f);
  for (int i = threadIdx.x; i < DD; i += 256)
    x1[row * DD + i] = resid[row * DD + i] + (p[i] - mean) * rstd * g[i] + bt[i];
}

// xn = bf16(LN(x1)*g + b)
__global__ __launch_bounds__(256) void ln_bf_kernel(
    const float* __restrict__ x1, const float* __restrict__ g,
    const float* __restrict__ bt, bf16* __restrict__ xn) {
  __shared__ float sbuf[256];
  const size_t row = blockIdx.x;
  const float* p = x1 + row * DD;
  float s = 0.f, sq = 0.f;
  for (int i = threadIdx.x; i < DD; i += 256) { const float v = p[i]; s += v; sq += v * v; }
  const float mean = block_reduce_sum(s, sbuf) * (1.0f / DD);
  const float var  = block_reduce_sum(sq, sbuf) * (1.0f / DD) - mean * mean;
  const float rstd = rsqrtf(var + 1e-5f);
  for (int i = threadIdx.x; i < DD; i += 256)
    xn[row * DD + i] = (bf16)((p[i] - mean) * rstd * g[i] + bt[i]);
}

// ====================================================================
// Launcher
// ====================================================================
static inline int eg(size_t n) { return (int)((n + 255) / 256); }

extern "C" void kernel_launch(void* const* d_in, const int* in_sizes, int n_in,
                              void* d_out, int out_size, void* d_ws, size_t ws_size,
                              hipStream_t stream) {
  const float* x         = (const float*)d_in[0];
  const float* cosp      = (const float*)d_in[1];
  const float* sinp      = (const float*)d_in[2];
  const float* gout      = (const float*)d_in[3];
  const float* lout      = (const float*)d_in[4];
  const float* wq        = (const float*)d_in[5];
  const float* wk        = (const float*)d_in[6];
  const float* wv        = (const float*)d_in[7];
  const float* wo        = (const float*)d_in[8];
  const float* pmem      = (const float*)d_in[9];
  const float* gate_w    = (const float*)d_in[10];
  const float* gate_b    = (const float*)d_in[11];
  const float* outproj_w = (const float*)d_in[12];
  const float* outproj_b = (const float*)d_in[13];
  const float* ln_mem_g  = (const float*)d_in[14];
  const float* ln_mem_b  = (const float*)d_in[15];
  const float* ln_mlp_g  = (const float*)d_in[16];
  const float* ln_mlp_b  = (const float*)d_in[17];
  const float* fc1_w     = (const float*)d_in[18];
  const float* fc1_b     = (const float*)d_in[19];
  const float* fc2_w     = (const float*)d_in[20];
  const float* fc2_b     = (const float*)d_in[21];
  float* outp = (float*)d_out;

  char* wsp = (char*)d_ws;
  auto alloc = [&](size_t bytes) -> char* {
    char* p = wsp;
    wsp += (bytes + 255) & ~(size_t)255;
    return p;
  };

  const int BLT = BB * LT;   // 8200
  const int BLL = BB * LL;   // 8192

  bf16* wq_bf   = (bf16*)alloc((size_t)DD * DD * 2);
  bf16* wk_bf   = (bf16*)alloc((size_t)DD * DD * 2);
  bf16* wv_bf   = (bf16*)alloc((size_t)DD * DD * 2);
  bf16* wo_bf   = (bf16*)alloc((size_t)DD * DD * 2);
  bf16* gate_bf = (bf16*)alloc((size_t)3 * DD * 3 * DD * 2);
  bf16* outp_bf = (bf16*)alloc((size_t)3 * DD * DD * 2);
  bf16* fc1_bf  = (bf16*)alloc((size_t)DD * 4 * DD * 2);
  bf16* fc2_bf  = (bf16*)alloc((size_t)4 * DD * DD * 2);
  bf16* xp_bf   = (bf16*)alloc((size_t)BLT * DD * 2);
  float* tmp0   = (float*)alloc((size_t)BLT * DD * 4);
  bf16* q_bf    = (bf16*)alloc((size_t)BLT * DD * 2);
  bf16* k_bf    = (bf16*)alloc((size_t)BLT * DD * 2);
  bf16* v_bf    = (bf16*)alloc((size_t)BB * HH * HD * LTP * 2);
  bf16* at_bf   = (bf16*)alloc((size_t)BLT * DD * 2);
  bf16* comb_bf = (bf16*)alloc((size_t)BLL * 3 * DD * 2);
  bf16* gated_bf= (bf16*)alloc((size_t)BLL * 3 * DD * 2);
  float* x1     = (float*)alloc((size_t)BLL * DD * 4);
  bf16* xn_bf   = (bf16*)alloc((size_t)BLL * DD * 2);
  bf16* h1_bf   = (bf16*)alloc((size_t)BLL * 4 * DD * 2);

  // ---- weight conversions f32 -> bf16 ----
  cvt_bf16_kernel<<<eg((size_t)DD * DD), 256, 0, stream>>>(wq, wq_bf, (size_t)DD * DD);
  cvt_bf16_kernel<<<eg((size_t)DD * DD), 256, 0, stream>>>(wk, wk_bf, (size_t)DD * DD);
  cvt_bf16_kernel<<<eg((size_t)DD * DD), 256, 0, stream>>>(wv, wv_bf, (size_t)DD * DD);
  cvt_bf16_kernel<<<eg((size_t)DD * DD), 256, 0, stream>>>(wo, wo_bf, (size_t)DD * DD);
  cvt_bf16_kernel<<<eg((size_t)9 * DD * DD), 256, 0, stream>>>(gate_w, gate_bf, (size_t)9 * DD * DD);
  cvt_bf16_kernel<<<eg((size_t)3 * DD * DD), 256, 0, stream>>>(outproj_w, outp_bf, (size_t)3 * DD * DD);
  cvt_bf16_kernel<<<eg((size_t)4 * DD * DD), 256, 0, stream>>>(fc1_w, fc1_bf, (size_t)4 * DD * DD);
  cvt_bf16_kernel<<<eg((size_t)4 * DD * DD), 256, 0, stream>>>(fc2_w, fc2_bf, (size_t)4 * DD * DD);

  // ---- xp = concat(pmem, x) ----
  build_xp_kernel<<<eg((size_t)BLT * DD), 256, 0, stream>>>(x, pmem, xp_bf);

  const dim3 blk(256);
  const dim3 gQKV(DD / 128, (BLT + 127) / 128);

  // ---- Q ----
  gemm_bf16_kernel<0><<<gQKV, blk, 0, stream>>>(xp_bf, wq_bf, nullptr, nullptr, tmp0, BLT, DD, DD);
  rope_kernel<<<eg((size_t)BLT * DD), 256, 0, stream>>>(tmp0, cosp, sinp, q_bf);
  // ---- K ----
  gemm_bf16_kernel<0><<<gQKV, blk, 0, stream>>>(xp_bf, wk_bf, nullptr, nullptr, tmp0, BLT, DD, DD);
  rope_kernel<<<eg((size_t)BLT * DD), 256, 0, stream>>>(tmp0, cosp, sinp, k_bf);
  // ---- V (transposed layout) ----
  gemm_bf16_kernel<0><<<gQKV, blk, 0, stream>>>(xp_bf, wv_bf, nullptr, nullptr, tmp0, BLT, DD, DD);
  vtrans_kernel<<<eg((size_t)BLT * DD), 256, 0, stream>>>(tmp0, v_bf);

  // ---- banded flash attention ----
  {
    const int tasks = BB * HH * QTILES;
    attn_kernel<<<(tasks + 7) / 8, 256, 0, stream>>>(q_bf, k_bf, v_bf, at_bf);
  }

  // ---- core = attn @ wo ----
  gemm_bf16_kernel<0><<<gQKV, blk, 0, stream>>>(at_bf, wo_bf, nullptr, nullptr, tmp0, BLT, DD, DD);

  // ---- combined = [global | local | core] ----
  pack_combined_kernel<<<eg((size_t)BLL * 3 * DD), 256, 0, stream>>>(gout, lout, tmp0, comb_bf);

  // ---- gated = combined * sigmoid(combined @ gate_w + gate_b) ----
  gemm_bf16_kernel<2><<<dim3(3 * DD / 128, BLL / 128), blk, 0, stream>>>(
      comb_bf, gate_bf, gate_b, comb_bf, gated_bf, BLL, 3 * DD, 3 * DD);

  // ---- memory_output = gated @ outproj + b -> tmp0 ----
  gemm_bf16_kernel<5><<<dim3(DD / 128, BLL / 128), blk, 0, stream>>>(
      gated_bf, outp_bf, outproj_b, nullptr, tmp0, BLL, DD, 3 * DD);

  // ---- x1 = x + LN(mo) ; xn = LN(x1) ----
  ln_add_kernel<<<BLL, 256, 0, stream>>>(tmp0, x, ln_mem_g, ln_mem_b, x1);
  ln_bf_kernel<<<BLL, 256, 0, stream>>>(x1, ln_mlp_g, ln_mlp_b, xn_bf);

  // ---- MLP ----
  gemm_bf16_kernel<3><<<dim3(4 * DD / 128, BLL / 128), blk, 0, stream>>>(
      xn_bf, fc1_bf, fc1_b, nullptr, h1_bf, BLL, 4 * DD, DD);
  gemm_bf16_kernel<4><<<dim3(DD / 128, BLL / 128), blk, 0, stream>>>(
      h1_bf, fc2_bf, fc2_b, x1, outp, BLL, DD, 4 * DD);

  (void)in_sizes; (void)n_in; (void)out_size; (void)ws_size;
}
